// FKAConvNetwork_78640851190287
// MI455X (gfx1250) — compile-verified, matrix operand order checked by reading, not run on hardware
//
#include <hip/hip_runtime.h>
#include <math.h>

// ---------------------------------------------------------------------------
// FKAConv network for MI455X (gfx1250, wave32).
// GEMM-shaped contractions -> v_wmma_f32_16x16x32_f16 (f32 accumulate),
// with pre-packed zero-padded f16 weights (branch-free b128 A loads) and
// Nn-templated immediate-offset B loads. Kernel-MLP + instance-norm stats
// on VALU with LDS reductions.
// ---------------------------------------------------------------------------

typedef __attribute__((ext_vector_type(16))) _Float16 v16h;
typedef __attribute__((ext_vector_type(8)))  _Float16 v8h;
typedef __attribute__((ext_vector_type(8)))  float    v8f;

static constexpr int   BB_  = 8;     // batch
static constexpr int   KNB  = 16;    // neighbors
static constexpr float EPSV = 1e-5f;

// ============================ stage A: pts / dw ============================
__global__ void stageA_kernel(const float* __restrict__ pos,
                              const float* __restrict__ sup,
                              const int*   __restrict__ idx,
                              float* __restrict__ pts,
                              float* __restrict__ dw,
                              const float* __restrict__ alpha,
                              const float* __restrict__ beta,
                              const float* __restrict__ nr,
                              int M, int Nn)
{
  int t = blockIdx.x * blockDim.x + threadIdx.x;
  if (t >= BB_ * M) return;
  int m = t % M, b = t / M;
  float al = alpha[0], be = beta[0], inr = 1.f / nr[0];
  const int* ib = idx + ((size_t)b * M + m) * KNB;
  float s0 = sup[((size_t)b * 3 + 0) * M + m];
  float s1 = sup[((size_t)b * 3 + 1) * M + m];
  float s2 = sup[((size_t)b * 3 + 2) * M + m];
  float dwk[KNB];
  float dws = 0.f;
  for (int k = 0; k < KNB; ++k) {
    int j = ib[k];
    float d0 = pos[((size_t)b * 3 + 0) * Nn + j] - s0;
    float d1 = pos[((size_t)b * 3 + 1) * Nn + j] - s1;
    float d2 = pos[((size_t)b * 3 + 2) * Nn + j] - s2;
    float dist = sqrtf(d0 * d0 + d1 * d1 + d2 * d2);
    pts[(((size_t)b * 3 + 0) * M + m) * KNB + k] = d0 * inr;
    pts[(((size_t)b * 3 + 1) * M + m) * KNB + k] = d1 * inr;
    pts[(((size_t)b * 3 + 2) * M + m) * KNB + k] = d2 * inr;
    float w = 1.f / (1.f + expf(al * dist - be));
    dwk[k] = w; dws += w;
  }
  dws = dws + ((dws == 0.f) ? 1.f : 0.f) + 1e-6f;
  float sc = (float)KNB / dws;
  for (int k = 0; k < KNB; ++k) dw[((size_t)b * M + m) * KNB + k] = dwk[k] * sc;
}

// ===================== fc1 (16x3) + inorm stat accumulate ===================
__global__ void fc1_kernel(const float* __restrict__ W,    // (16,3)
                           const float* __restrict__ pts,  // (B,3,M,K)
                           float* __restrict__ out,        // (B,16,M,K)
                           float* __restrict__ stats,      // (B*16,2)
                           int M)
{
  __shared__ float sh0[256];
  __shared__ float sh1[256];
  int tid = threadIdx.x;
  int m = blockIdx.x * blockDim.x + tid;
  int s = blockIdx.y, b = blockIdx.z;
  float w0 = W[s * 3 + 0], w1 = W[s * 3 + 1], w2 = W[s * 3 + 2];
  float ls = 0.f, lss = 0.f;
  if (m < M) {
    const float* p0 = pts + (((size_t)b * 3 + 0) * M + m) * KNB;
    const float* p1 = pts + (((size_t)b * 3 + 1) * M + m) * KNB;
    const float* p2 = pts + (((size_t)b * 3 + 2) * M + m) * KNB;
    float* o = out + (((size_t)b * 16 + s) * M + m) * KNB;
    #pragma unroll
    for (int k = 0; k < KNB; ++k) {
      float v = w0 * p0[k] + w1 * p1[k] + w2 * p2[k];
      o[k] = v; ls += v; lss += v * v;
    }
  }
  sh0[tid] = ls; sh1[tid] = lss;
  __syncthreads();
  for (int off = 128; off > 0; off >>= 1) {
    if (tid < off) { sh0[tid] += sh0[tid + off]; sh1[tid] += sh1[tid + off]; }
    __syncthreads();
  }
  if (tid == 0) {
    atomicAdd(&stats[((size_t)b * 16 + s) * 2 + 0], sh0[0]);
    atomicAdd(&stats[((size_t)b * 16 + s) * 2 + 1], sh1[0]);
  }
}

// =========== fc2/fc3 (16x32) [+ stats] [+ relu*dw] over (B,32,M,K) ==========
__global__ void fcmid_kernel(const float* __restrict__ W,   // (16,32)
                             const float* __restrict__ in,  // (B,32,M,K)
                             float* __restrict__ out,       // (B,16,M,K)
                             float* __restrict__ stats,     // or null
                             const float* __restrict__ dw,  // or null
                             int M, int relu_dw)
{
  __shared__ float sh0[256];
  __shared__ float sh1[256];
  int tid = threadIdx.x;
  int m = blockIdx.x * blockDim.x + tid;
  int s = blockIdx.y, b = blockIdx.z;
  float wr[32];
  #pragma unroll
  for (int c = 0; c < 32; ++c) wr[c] = W[s * 32 + c];
  float ls = 0.f, lss = 0.f;
  if (m < M) {
    const float* base = in + (size_t)b * 32 * M * KNB + (size_t)m * KNB;
    size_t cst = (size_t)M * KNB;
    float* o = out + (((size_t)b * 16 + s) * M + m) * KNB;
    const float* dwp = dw ? (dw + ((size_t)b * M + m) * KNB) : nullptr;
    for (int k = 0; k < KNB; ++k) {
      float acc = 0.f;
      #pragma unroll
      for (int c = 0; c < 32; ++c) acc += wr[c] * base[(size_t)c * cst + k];
      if (relu_dw) acc = fmaxf(acc, 0.f) * dwp[k];
      o[k] = acc; ls += acc; lss += acc * acc;
    }
  }
  if (stats) {
    sh0[tid] = ls; sh1[tid] = lss;
    __syncthreads();
    for (int off = 128; off > 0; off >>= 1) {
      if (tid < off) { sh0[tid] += sh0[tid + off]; sh1[tid] += sh1[tid + off]; }
      __syncthreads();
    }
    if (tid == 0) {
      atomicAdd(&stats[((size_t)b * 16 + s) * 2 + 0], sh0[0]);
      atomicAdd(&stats[((size_t)b * 16 + s) * 2 + 1], sh1[0]);
    }
  }
}

// ======================== inorm: sums -> mean/invstd ========================
__global__ void inorm_fin_kernel(const float* __restrict__ stats,
                                 float* __restrict__ mi, float cnt)
{
  int t = blockIdx.x * blockDim.x + threadIdx.x;
  if (t >= BB_ * 16) return;
  float su = stats[t * 2 + 0], sq = stats[t * 2 + 1];
  float mean = su / cnt;
  float var = sq / cnt - mean * mean;
  mi[t * 2 + 0] = mean;
  mi[t * 2 + 1] = rsqrtf(var + EPSV);
}

// ========= normalize + relu, and concat broadcast max_k(m*dw) (ch 16..31) ===
__global__ void mid_kernel(const float* __restrict__ t,   // (B,16,M,K)
                           const float* __restrict__ mi,  // (B*16,2)
                           const float* __restrict__ g, const float* __restrict__ bta,
                           const float* __restrict__ dw,  // (B,M,K)
                           float* __restrict__ mcat,      // (B,32,M,K)
                           int M)
{
  int tt = blockIdx.x * blockDim.x + threadIdx.x;
  if (tt >= BB_ * 16 * M) return;
  int m = tt % M, s = (tt / M) & 15, b = tt / (16 * M);
  float mean = mi[((size_t)b * 16 + s) * 2 + 0];
  float is   = mi[((size_t)b * 16 + s) * 2 + 1];
  float gg = g[s], bb2 = bta[s];
  const float* ti  = t  + (((size_t)b * 16 + s) * M + m) * KNB;
  const float* dwp = dw + ((size_t)b * M + m) * KNB;
  float* lo = mcat + (((size_t)b * 32 + s)      * M + m) * KNB;
  float* hi = mcat + (((size_t)b * 32 + 16 + s) * M + m) * KNB;
  float mx = -3.4e38f;
  #pragma unroll
  for (int k = 0; k < KNB; ++k) {
    float v = (ti[k] - mean) * is * gg + bb2;
    v = fmaxf(v, 0.f);
    lo[k] = v;
    mx = fmaxf(mx, v * dwp[k]);
  }
  #pragma unroll
  for (int k = 0; k < KNB; ++k) hi[k] = mx;
}

// ===== feat[b, c*16+s, m] = sum_k h[b,c,idx[b,m,k]] * m3[b,s,m,k] (gathered) =
// feat leading dim is CSp (zero-padded to a multiple of 32 rows).
__global__ void feat_kernel(const float* __restrict__ h,   // (B,C,Nn)
                            const int*   __restrict__ idx, // (B,M,K)
                            const float* __restrict__ m3,  // (B,16,M,K)
                            float* __restrict__ feat,      // (B,CSp,M)
                            int C, int CSp, int M, int Nn)
{
  int t = blockIdx.x * blockDim.x + threadIdx.x;
  if (t >= BB_ * C * M) return;
  int m = t % M, c = (t / M) % C, b = t / (M * C);
  const int* ib = idx + ((size_t)b * M + m) * KNB;
  const float* hb = h + ((size_t)b * C + c) * Nn;
  const float* mb = m3 + (size_t)b * 16 * M * KNB + (size_t)m * KNB;
  size_t sst = (size_t)M * KNB;
  float acc[16];
  #pragma unroll
  for (int s = 0; s < 16; ++s) acc[s] = 0.f;
  for (int k = 0; k < KNB; ++k) {
    float xk = hb[ib[k]];
    #pragma unroll
    for (int s = 0; s < 16; ++s) acc[s] += xk * mb[(size_t)s * sst + k];
  }
  float* fb = feat + (((size_t)b * CSp + (size_t)c * 16) * M + m);
  #pragma unroll
  for (int s = 0; s < 16; ++s) fb[(size_t)s * M] = acc[s];
}

// ================= weight pack: f32 (OxCd) -> f16 zero-padded (OpxCdp) ======
__global__ void pack_w_kernel(const float* __restrict__ W, _Float16* __restrict__ Wp,
                              int O, int Cd, int Cdp, int total)
{
  int t = blockIdx.x * blockDim.x + threadIdx.x;
  if (t >= total) return;
  int c = t % Cdp, o = t / Cdp;
  float v = (o < O && c < Cd) ? W[(size_t)o * Cd + c] : 0.f;
  Wp[t] = (_Float16)v;
}

// ============================== WMMA GEMM ==================================
// Y[b,o,n] = act( bn( Wp(OpxCdp,f16) @ X[b](CdpxNN) + bias ) [+ res] )
// One wave per 16x16 tile. A: two unconditional b128 loads per K-step from the
// packed f16 weights (ISA A layout: halves 0..7 = K kb..kb+7, 8..15 = kb+16..23,
// kb = 8*(lane>=16)). B: 16 immediate-offset b32 loads (NN compile-time).
template <int NN>
__device__ __forceinline__ v8f gemm_step(const _Float16* __restrict__ wrow,
                                         const float* __restrict__ bcol,
                                         int c0, v8f acc)
{
  union { v16h v; v8h h[2]; } A;
  A.h[0] = *(const v8h*)(wrow + c0);
  A.h[1] = *(const v8h*)(wrow + c0 + 16);
  union { v16h v; _Float16 e[16]; } Bf;
  const float* bp = bcol + (size_t)c0 * NN;
  #pragma unroll
  for (int e = 0; e < 16; ++e) {
    int off = (e < 8) ? e : (e + 8);     // compile-time * NN -> immediate offset
    Bf.e[e] = (_Float16)bp[(size_t)off * NN];
  }
  return __builtin_amdgcn_wmma_f32_16x16x32_f16(
      false, A.v, false, Bf.v, (short)0, acc, false, false);
}

template <int NN>
__global__ void wmma_gemm_kernel(const _Float16* __restrict__ Wp,
                                 const float* __restrict__ X,
                                 float* __restrict__ Y,
                                 const float* __restrict__ bias,
                                 const float* __restrict__ bng,
                                 const float* __restrict__ bnb,
                                 const float* __restrict__ res,
                                 int O, int Cdp, int act)
{
  int lane = threadIdx.x & 31;
  int nt = blockIdx.x * 16;
  int ot = blockIdx.y * 16;
  int b  = blockIdx.z;
  int l  = lane & 15;
  int hi = lane >> 4;
  int kb = hi * 8;
  int row = ot + l;
  int col = nt + l;
  const _Float16* wrow = Wp + (size_t)row * Cdp + kb;
  const float*    bcol = X + (size_t)b * Cdp * NN + (size_t)kb * NN + col;
  union { v8f v; float f[8]; } a0, a1;
  #pragma unroll
  for (int i = 0; i < 8; ++i) { a0.f[i] = 0.f; a1.f[i] = 0.f; }
  int c0 = 0;
  for (; c0 + 64 <= Cdp; c0 += 64) {       // 2 independent acc chains
    __builtin_prefetch(bcol + (size_t)(c0 + 64) * NN, 0, 1);
    a0.v = gemm_step<NN>(wrow, bcol, c0,      a0.v);
    a1.v = gemm_step<NN>(wrow, bcol, c0 + 32, a1.v);
  }
  if (c0 < Cdp) a0.v = gemm_step<NN>(wrow, bcol, c0, a0.v);

  const float bnscale = rsqrtf(1.f + EPSV);   // bn1d: x/sqrt(1+eps)*g + b
  #pragma unroll
  for (int r = 0; r < 8; ++r) {
    int ro = ot + r + hi * 8;   // C/D layout: lanes<16 rows 0..7, lanes>=16 rows 8..15
    if (ro >= O) continue;
    float v = a0.f[r] + a1.f[r];
    if (bias) v += bias[ro];
    if (bng)  v = v * (bng[ro] * bnscale) + bnb[ro];
    if (res)  v += res[((size_t)b * O + ro) * NN + col];
    if (act == 1)      v = fmaxf(v, 0.f);
    else if (act == 2) v = (v >= 0.f) ? v : 0.1f * v;
    Y[((size_t)b * O + ro) * NN + col] = v;
  }
}

template <int NN>
static void launch_wmma_gemm(const _Float16* Wp, const float* X, float* Y,
                             const float* bias, const float* g, const float* bb,
                             const float* res, int O, int Cdp, int act,
                             hipStream_t stream)
{
  dim3 grid(NN / 16, (O + 15) / 16, BB_);
  wmma_gemm_kernel<NN><<<grid, dim3(32), 0, stream>>>(Wp, X, Y, bias, g, bb,
                                                      res, O, Cdp, act);
}

// ============================ shortcut max-pool ============================
__global__ void maxpool_kernel(const float* __restrict__ in,  // (B,C,Nn)
                               const int*   __restrict__ idx, // (B,M,K)
                               float* __restrict__ out,       // (B,C,M)
                               int C, int Nn, int M)
{
  int t = blockIdx.x * blockDim.x + threadIdx.x;
  if (t >= BB_ * C * M) return;
  int m = t % M, c = (t / M) % C, b = t / (M * C);
  const int* ib = idx + ((size_t)b * M + m) * KNB;
  const float* hb = in + ((size_t)b * C + c) * Nn;
  float mx = -3.4e38f;
  #pragma unroll
  for (int k = 0; k < KNB; ++k) mx = fmaxf(mx, hb[ib[k]]);
  out[((size_t)b * C + c) * M + m] = mx;
}

// ============================ final mean over points ========================
__global__ void mean_kernel(const float* __restrict__ y, float* __restrict__ out)
{
  int t = blockIdx.x * blockDim.x + threadIdx.x;
  if (t >= BB_ * 40) return;
  const float* p = y + (size_t)t * 16;
  float s = 0.f;
  #pragma unroll
  for (int i = 0; i < 16; ++i) s += p[i];
  out[t] = s * (1.f / 16.f);
}

// ===========================================================================
extern "C" void kernel_launch(void* const* d_in, const int* in_sizes, int n_in,
                              void* d_out, int out_size, void* d_ws, size_t ws_size,
                              hipStream_t stream)
{
  (void)n_in; (void)out_size; (void)ws_size;

  const float* x    = (const float*)d_in[0];
  const float* pos  = (const float*)d_in[1];
  const float* sup1 = (const float*)d_in[2];
  const float* sup2 = (const float*)d_in[3];
  const float* sup3 = (const float*)d_in[4];
  const float* sup4 = (const float*)d_in[5];

  // setup_inputs() dict insertion order is {x,pos,sup1..4,params,ids...}; some
  // harnesses flatten the reference arg order (ids before params). Detect via
  // in_sizes[6]: ids00 = 8*4096*16 = 524288, first params leaf (cv0.cv) = 3072.
  const int NPARAM = 220;
  int idbase, pbase;
  if (in_sizes[6] == BB_ * 4096 * KNB) { idbase = 6; pbase = 15; }
  else                                 { pbase = 6; idbase = 6 + NPARAM; }

  const int* ids00 = (const int*)d_in[idbase + 0];
  const int* ids01 = (const int*)d_in[idbase + 1];
  const int* ids11 = (const int*)d_in[idbase + 2];
  const int* ids12 = (const int*)d_in[idbase + 3];
  const int* ids22 = (const int*)d_in[idbase + 4];
  const int* ids23 = (const int*)d_in[idbase + 5];
  const int* ids33 = (const int*)d_in[idbase + 6];
  const int* ids34 = (const int*)d_in[idbase + 7];
  const int* ids44 = (const int*)d_in[idbase + 8];

  int pi = pbase;
  auto F = [&]() -> const float* { return (const float*)d_in[pi++]; };
  struct FkaP { const float *cv,*fc1,*fc2,*fc3,*i1g,*i1b,*i2g,*i2b,*al,*be,*nr; };
  auto loadFka = [&]() -> FkaP {
    FkaP p; p.cv=F(); p.fc1=F(); p.fc2=F(); p.fc3=F();
    p.i1g=F(); p.i1b=F(); p.i2g=F(); p.i2b=F(); p.al=F(); p.be=F(); p.nr=F();
    return p;
  };
  struct ResP { const float *c0w,*c0b,*b0g,*b0b; FkaP fka;
                const float *b1g,*b1b,*c2w,*c2b,*b2g,*b2b,*shw,*shb,*bsg,*bsb; bool hs; };
  auto loadRes = [&](bool hs) -> ResP {
    ResP r; r.c0w=F(); r.c0b=F(); r.b0g=F(); r.b0b=F(); r.fka=loadFka();
    r.b1g=F(); r.b1b=F(); r.c2w=F(); r.c2b=F(); r.b2g=F(); r.b2b=F();
    r.hs=hs; r.shw=r.shb=r.bsg=r.bsb=nullptr;
    if (hs) { r.shw=F(); r.shb=F(); r.bsg=F(); r.bsb=F(); }
    return r;
  };
  FkaP cv0 = loadFka();
  const float* bn0g = F(); const float* bn0b = F();
  ResP b01 = loadRes(false), b10 = loadRes(true),  b11 = loadRes(false),
       b20 = loadRes(true),  b21 = loadRes(false), b30 = loadRes(true),
       b31 = loadRes(false), b40 = loadRes(true),  b41 = loadRes(false);
  const float* fcow = F(); const float* fcob = F();

  // ------------------------- workspace arena (floats) ----------------------
  float* w = (float*)d_ws;
  auto carve = [&](size_t n) -> float* { float* p = w; w += n; return p; };
  float* bufA  = carve(2097152);   // act ping (B*64*4096 max)
  float* bufB  = carve(2097152);   // act pong
  float* bufH  = carve(1048576);   // h1 = lrelu(bn(cv0@x)); also fcout out
  float* bufF  = carve(1048576);   // fkaconv output (mid x M)
  float* bufS  = carve(4194304);   // shortcut at N points (B*128*4096 max)
  float* bufS2 = carve(1048576);   // pooled shortcut
  float* pts   = carve(1572864);   // (B,3,M,K)
  float* dwb   = carve(524288);    // (B,M,K)
  float* tb    = carve(8388608);   // (B,16,M,K)   t1 / t2 / m3 (reused)
  float* mc    = carve(16777216);  // (B,32,M,K)   mcat; then feat (B,CSp,M)
  float* st    = carve(256);       // raw stats (B*16 x {sum,sumsq})
  float* stf   = carve(256);       // finalized {mean, invstd}
  _Float16* wpack = (_Float16*)carve(2097152); // packed f16 weights (<= 512x8192)

  auto gemm = [&](const float* W_, const float* X_, float* Y_,
                  const float* bias, const float* g, const float* bb,
                  const float* res, int O, int Cd, int Nn, int act) {
    int Op  = (O + 15) & ~15;
    int Cdp = (Cd + 31) & ~31;
    int tot = Op * Cdp;
    pack_w_kernel<<<(tot + 255) / 256, 256, 0, stream>>>(W_, wpack, O, Cd, Cdp, tot);
    switch (Nn) {
      case 4096: launch_wmma_gemm<4096>(wpack, X_, Y_, bias, g, bb, res, O, Cdp, act, stream); break;
      case 1024: launch_wmma_gemm<1024>(wpack, X_, Y_, bias, g, bb, res, O, Cdp, act, stream); break;
      case 256:  launch_wmma_gemm< 256>(wpack, X_, Y_, bias, g, bb, res, O, Cdp, act, stream); break;
      case 64:   launch_wmma_gemm<  64>(wpack, X_, Y_, bias, g, bb, res, O, Cdp, act, stream); break;
      default:   launch_wmma_gemm<  16>(wpack, X_, Y_, bias, g, bb, res, O, Cdp, act, stream); break;
    }
  };

  auto fka = [&](const float* h, int C, const float* pp, const float* su,
                 const int* idx, int M, int Nn, const FkaP& P,
                 float* out, int O, const float* g, const float* bb, int act) {
    int totM = BB_ * M;
    stageA_kernel<<<(totM + 127) / 128, 128, 0, stream>>>(
        pp, su, idx, pts, dwb, P.al, P.be, P.nr, M, Nn);
    float cnt = (float)((size_t)M * KNB);
    dim3 gfc((M + 255) / 256, 16, BB_);
    int totSM = BB_ * 16 * M;

    hipMemsetAsync(st, 0, 256 * sizeof(float), stream);
    fc1_kernel<<<gfc, 256, 0, stream>>>(P.fc1, pts, tb, st, M);
    inorm_fin_kernel<<<1, 128, 0, stream>>>(st, stf, cnt);
    mid_kernel<<<(totSM + 255) / 256, 256, 0, stream>>>(tb, stf, P.i1g, P.i1b, dwb, mc, M);

    hipMemsetAsync(st, 0, 256 * sizeof(float), stream);
    fcmid_kernel<<<gfc, 256, 0, stream>>>(P.fc2, mc, tb, st, nullptr, M, 0);
    inorm_fin_kernel<<<1, 128, 0, stream>>>(st, stf, cnt);
    mid_kernel<<<(totSM + 255) / 256, 256, 0, stream>>>(tb, stf, P.i2g, P.i2b, dwb, mc, M);

    fcmid_kernel<<<gfc, 256, 0, stream>>>(P.fc3, mc, tb, nullptr, dwb, M, 1);

    int CS = C * 16, CSp = (CS + 31) & ~31;
    if (CSp != CS)  // only first layer (C=3): zero the padded feat rows
      hipMemsetAsync(mc, 0, (size_t)BB_ * CSp * M * sizeof(float), stream);
    int totF = BB_ * C * M;
    feat_kernel<<<(totF + 255) / 256, 256, 0, stream>>>(h, idx, tb, mc, C, CSp, M, Nn);
    gemm(P.cv, mc, out, nullptr, g, bb, nullptr, O, CS, M, act);
  };

  auto resblock = [&](const ResP& R, const float* cur, int cin, int cout,
                      const float* pp, const float* su, const int* idx,
                      int Nn, int M, float* out) {
    int mid = cin / 2;
    const float* resptr;
    if (R.hs) {
      gemm(R.shw, cur, bufS, R.shb, R.bsg, R.bsb, nullptr, cout, cin, Nn, 0);
      if (M != Nn) {
        int tot = BB_ * cout * M;
        maxpool_kernel<<<(tot + 255) / 256, 256, 0, stream>>>(bufS, idx, bufS2, cout, Nn, M);
        resptr = bufS2;
      } else resptr = bufS;
    } else resptr = cur;                         // cin==cout and M==Nn here
    gemm(R.c0w, cur, bufH, R.c0b, R.b0g, R.b0b, nullptr, mid, cin, Nn, 2);
    fka(bufH, mid, pp, su, idx, M, Nn, R.fka, bufF, mid, R.b1g, R.b1b, 2);
    gemm(R.c2w, bufF, out, R.c2b, R.b2g, R.b2b, resptr, cout, mid, M, 2);
  };

  // -------------------------------- network --------------------------------
  fka(x, 3, pos, pos, ids00, 4096, 4096, cv0, bufA, 64, bn0g, bn0b, 1 /*relu*/);
  resblock(b01, bufA,   64,   64, pos,  pos,  ids00, 4096, 4096, bufB);
  resblock(b10, bufB,   64,  128, pos,  sup1, ids01, 4096, 1024, bufA);
  resblock(b11, bufA,  128,  128, sup1, sup1, ids11, 1024, 1024, bufB);
  resblock(b20, bufB,  128,  256, sup1, sup2, ids12, 1024,  256, bufA);
  resblock(b21, bufA,  256,  256, sup2, sup2, ids22,  256,  256, bufB);
  resblock(b30, bufB,  256,  512, sup2, sup3, ids23,  256,   64, bufA);
  resblock(b31, bufA,  512,  512, sup3, sup3, ids33,   64,   64, bufB);
  resblock(b40, bufB,  512, 1024, sup3, sup4, ids34,   64,   16, bufA);
  resblock(b41, bufA, 1024, 1024, sup4, sup4, ids44,   16,   16, bufB);

  gemm(fcow, bufB, bufH, fcob, nullptr, nullptr, nullptr, 40, 1024, 16, 0);
  mean_kernel<<<(BB_ * 40 + 127) / 128, 128, 0, stream>>>(bufH, (float*)d_out);
}
